// BidirectionalGRU_39127152067125
// MI455X (gfx1250) — matrix operand order repaired and checked by reading, not compiled, not run on hardware
//
#include <hip/hip_runtime.h>
#include <hip/hip_bf16.h>
#include <math.h>

// Problem constants (from reference)
#define B_    8
#define S_    1024
#define D_    1024
#define H_    1024
#define M_    (B_ * S_)      // 8192 rows
#define G3_   (3 * H_)       // 3072 gate cols
#define FFN_  2816

#if defined(__AMDGCN__) && __has_builtin(__builtin_amdgcn_global_load_async_to_lds_b128)
#define HAVE_ASYNC_LDS 1
#else
#define HAVE_ASYNC_LDS 0
#endif

typedef __attribute__((ext_vector_type(16))) __bf16 v16bf;
typedef __attribute__((ext_vector_type(8)))  float  v8f;
typedef int v4i_ __attribute__((vector_size(16)));   // matches builtin param pointee

union FragU {
  uint4          q[2];
  v16bf          v;
  unsigned short s[16];
  unsigned int   d[8];
};

// ---------------------------------------------------------------- helpers ---
__device__ __forceinline__ unsigned short f2bf(float f) {
  unsigned int u = __float_as_uint(f);
  u += 0x7fffu + ((u >> 16) & 1u);            // round-to-nearest-even
  return (unsigned short)(u >> 16);
}

// 16-bit A-matrix 16x32 layout (ISA 7.12.2): lane L row M=L%16, half=L/16.
// VGPR v (elements 2v,2v+1): v<4 -> K=8*half+2v ; v>=4 -> K=16+8*half+2(v-4)
__device__ __forceinline__ int a_k_of(int lane, int e) {
  int half = lane >> 4;
  int v    = e >> 1;
  int klo  = (v < 4) ? (8 * half + 2 * v) : (16 + 8 * half + 2 * (v - 4));
  return klo + (e & 1);
}

// B-matrix 32x16 (KxN): lane L col N=L%16; same K order as A (a_k_of)
__device__ __forceinline__ int b_k_of(int lane, int e) {
  return a_k_of(lane, e);
}

__device__ __forceinline__ float sigmoidf_(float x) {
  return 1.0f / (1.0f + __expf(-x));
}

// ---------------------------------------------------------- pack kernels ---
// A[M,K] f32 row-major -> fragment buffer [Mt][Kt][32 lanes][16 elems] bf16
__global__ __launch_bounds__(256) void pack_a_bf16(const float* __restrict__ A,
                                                   unsigned short* __restrict__ out,
                                                   int M, int K, int Kt, int total) {
  int gid = blockIdx.x * blockDim.x + threadIdx.x;
  if (gid >= total) return;
  int lane = gid & 31;
  int tile = gid >> 5;                 // mt*Kt + kt
  int kt   = tile % Kt;
  int mt   = tile / Kt;
  int m    = mt * 16 + (lane & 15);
  FragU f;
#pragma unroll
  for (int e = 0; e < 16; ++e) {
    int   k = kt * 32 + a_k_of(lane, e);
    float v = (m < M) ? A[(size_t)m * K + k] : 0.0f;
    f.s[e]  = f2bf(v);
  }
  uint4* o = (uint4*)(out + (size_t)tile * 512 + lane * 16);
  o[0] = f.q[0];
  o[1] = f.q[1];
}

// Fused SwiGLU + A-pack: val = silu(h1)*h3, packed directly into fragments.
__global__ __launch_bounds__(256) void pack_a_swiglu(const float* __restrict__ h1,
                                                     const float* __restrict__ h3,
                                                     unsigned short* __restrict__ out,
                                                     int M, int K, int Kt, int total) {
  int gid = blockIdx.x * blockDim.x + threadIdx.x;
  if (gid >= total) return;
  int lane = gid & 31;
  int tile = gid >> 5;
  int kt   = tile % Kt;
  int mt   = tile / Kt;
  int m    = mt * 16 + (lane & 15);
  FragU f;
#pragma unroll
  for (int e = 0; e < 16; ++e) {
    int    k = kt * 32 + a_k_of(lane, e);
    float  v = 0.0f;
    if (m < M) {
      size_t idx = (size_t)m * K + k;
      float  a   = h1[idx];
      v          = (a * sigmoidf_(a)) * h3[idx];
    }
    f.s[e] = f2bf(v);
  }
  uint4* o = (uint4*)(out + (size_t)tile * 512 + lane * 16);
  o[0] = f.q[0];
  o[1] = f.q[1];
}

// W[N,K] f32 row-major -> fragment buffer [Nt][Kt][32 lanes][16 elems] bf16
__global__ __launch_bounds__(256) void pack_b_bf16(const float* __restrict__ W,
                                                   unsigned short* __restrict__ out,
                                                   int N, int K, int Kt, int total) {
  int gid = blockIdx.x * blockDim.x + threadIdx.x;
  if (gid >= total) return;
  int lane = gid & 31;
  int tile = gid >> 5;                 // nt*Kt + kt
  int kt   = tile % Kt;
  int nt   = tile / Kt;
  int n    = nt * 16 + (lane & 15);
  FragU f;
#pragma unroll
  for (int e = 0; e < 16; ++e) {
    int   k = kt * 32 + b_k_of(lane, e);
    float v = (n < N) ? W[(size_t)n * K + k] : 0.0f;
    f.s[e]  = f2bf(v);
  }
  uint4* o = (uint4*)(out + (size_t)tile * 512 + lane * 16);
  o[0] = f.q[0];
  o[1] = f.q[1];
}

// ----------------------------------------------------------- rmsnorm ------
__global__ __launch_bounds__(256) void rmsnorm_kernel(const float* __restrict__ x,
                                                      const float* __restrict__ w,
                                                      float* __restrict__ out, int D) {
  __shared__ float red[256];
  int          row = blockIdx.x;
  const float* xr  = x + (size_t)row * D;
  float        ss  = 0.0f;
  for (int i = threadIdx.x; i < D; i += 256) {
    float v = xr[i];
    ss += v * v;
  }
  red[threadIdx.x] = ss;
  __syncthreads();
  for (int s = 128; s > 0; s >>= 1) {
    if (threadIdx.x < s) red[threadIdx.x] += red[threadIdx.x + s];
    __syncthreads();
  }
  float scale = __frsqrt_rn(red[0] / (float)D + 1e-5f);
  for (int i = threadIdx.x; i < D; i += 256)
    out[(size_t)row * D + i] = xr[i] * scale * w[i];
}

// ------------------------------------------------------- WMMA GEMM --------
// C[M,N] = A @ B^T (+bias) (+resid). Register-blocked: each wave computes a
// 32x64 macro-tile (2 M-tiles x 4 N-tiles) -> A frags reused 4x, B frags 2x.
__global__ __launch_bounds__(256) void wmma_gemm_kernel(
    const unsigned short* __restrict__ Ap, const unsigned short* __restrict__ Bp,
    const float* __restrict__ bias, const float* __restrict__ resid,
    float* __restrict__ C, int Mt2, int Nt4, int Kt, int M, int N) {
  int wave = threadIdx.x >> 5;
  int lane = threadIdx.x & 31;
  int tile = blockIdx.x * 8 + wave;
  if (tile >= Mt2 * Nt4) return;                   // wave-uniform
  int nt0 = (tile % Nt4) * 4;
  int mt0 = (tile / Nt4) * 2;

  const uint4* ap0 = (const uint4*)(Ap + ((size_t)mt0 * Kt) * 512 + lane * 16);
  const uint4* ap1 = (const uint4*)(Ap + ((size_t)(mt0 + 1) * Kt) * 512 + lane * 16);
  const uint4* bp0 = (const uint4*)(Bp + ((size_t)(nt0 + 0) * Kt) * 512 + lane * 16);
  const uint4* bp1 = (const uint4*)(Bp + ((size_t)(nt0 + 1) * Kt) * 512 + lane * 16);
  const uint4* bp2 = (const uint4*)(Bp + ((size_t)(nt0 + 2) * Kt) * 512 + lane * 16);
  const uint4* bp3 = (const uint4*)(Bp + ((size_t)(nt0 + 3) * Kt) * 512 + lane * 16);

  v8f acc[2][4] = {};
  for (int kt = 0; kt < Kt; ++kt) {
    FragU a0, a1;
    a0.q[0] = ap0[0]; a0.q[1] = ap0[1];
    a1.q[0] = ap1[0]; a1.q[1] = ap1[1];
    if (kt + 8 < Kt) {                 // stream-prefetch B fragments
      __builtin_prefetch((const char*)bp0 + 8 * 1024, 0, 0);
      __builtin_prefetch((const char*)bp2 + 8 * 1024, 0, 0);
    }
    FragU b;
    b.q[0] = bp0[0]; b.q[1] = bp0[1];
    acc[0][0] = __builtin_amdgcn_wmma_f32_16x16x32_bf16(false, a0.v, false, b.v, (short)0, acc[0][0], false, false);
    acc[1][0] = __builtin_amdgcn_wmma_f32_16x16x32_bf16(false, a1.v, false, b.v, (short)0, acc[1][0], false, false);
    b.q[0] = bp1[0]; b.q[1] = bp1[1];
    acc[0][1] = __builtin_amdgcn_wmma_f32_16x16x32_bf16(false, a0.v, false, b.v, (short)0, acc[0][1], false, false);
    acc[1][1] = __builtin_amdgcn_wmma_f32_16x16x32_bf16(false, a1.v, false, b.v, (short)0, acc[1][1], false, false);
    b.q[0] = bp2[0]; b.q[1] = bp2[1];
    acc[0][2] = __builtin_amdgcn_wmma_f32_16x16x32_bf16(false, a0.v, false, b.v, (short)0, acc[0][2], false, false);
    acc[1][2] = __builtin_amdgcn_wmma_f32_16x16x32_bf16(false, a1.v, false, b.v, (short)0, acc[1][2], false, false);
    b.q[0] = bp3[0]; b.q[1] = bp3[1];
    acc[0][3] = __builtin_amdgcn_wmma_f32_16x16x32_bf16(false, a0.v, false, b.v, (short)0, acc[0][3], false, false);
    acc[1][3] = __builtin_amdgcn_wmma_f32_16x16x32_bf16(false, a1.v, false, b.v, (short)0, acc[1][3], false, false);
    ap0 += 64; ap1 += 64; bp0 += 64; bp1 += 64; bp2 += 64; bp3 += 64;
  }

#pragma unroll
  for (int j = 0; j < 4; ++j) {
    int   col  = (nt0 + j) * 16 + (lane & 15);
    float badd = bias ? bias[col] : 0.0f;
#pragma unroll
    for (int mi = 0; mi < 2; ++mi) {
      int rbase = (mt0 + mi) * 16 + ((lane >> 4) << 3);
#pragma unroll
      for (int i = 0; i < 8; ++i) {
        int row = rbase + i;
        if (row < M) {
          float v = acc[mi][j][i] + badd;
          if (resid) v += resid[(size_t)row * N + col];
          C[(size_t)row * N + col] = v;
        }
      }
    }
  }
}

// -------------------------------------------------------- GRU scan --------
// Persistent kernel: 8 blocks x 8 waves = 64 waves; wave owns 16 h-columns.
// h kept in f32 (hbuf) for recurrence precision + bf16 shadow (hb16) that is
// DMA-staged into LDS each step via GLOBAL_LOAD_ASYNC_TO_LDS (ASYNCcnt).
__global__ __launch_bounds__(256, 1) void gru_scan_kernel(
    const float* __restrict__ xg,            // [B*S, 3H], row = b*S + t
    const unsigned short* __restrict__ whh,  // packed [192][32][512]
    const float* __restrict__ bhh,           // [3H]
    float* __restrict__ hbuf,                // [2][B][H] f32
    unsigned short* __restrict__ hb16,       // [2][B][H] bf16 shadow
    float* __restrict__ out,                 // [B,S,2H]
    unsigned int* __restrict__ counter, int reverse, int col_off) {
  const int H = H_, Kt = 32, nblk = gridDim.x;
  __shared__ unsigned short hls[16 * H_];    // 32 KB, rows 8..15 zero-padded

  int wave = threadIdx.x >> 5;
  int lane = threadIdx.x & 31;
  int jt   = blockIdx.x * 8 + wave;          // 0..63
  int j0   = jt * 16;
  int ln15 = lane & 15;
  int half = lane >> 4;

  // zero pad rows 8..15 once (never overwritten afterwards)
  for (int idx = 8 * H + (int)threadIdx.x; idx < 16 * H; idx += 256) hls[idx] = 0;

  for (int t = 0; t < S_; ++t) {
    int                   te    = reverse ? (S_ - 1 - t) : t;
    const float*          hcur  = hbuf + (size_t)(t & 1) * B_ * H;
    float*                hnxt  = hbuf + (size_t)((t + 1) & 1) * B_ * H;
    const unsigned short* h16c  = hb16 + (size_t)(t & 1) * B_ * H;
    unsigned short*       h16n  = hb16 + (size_t)((t + 1) & 1) * B_ * H;

    // stage h rows 0..7 (bf16, 16 KB) into LDS
    if (t == 0) {
      uint4  z  = {0u, 0u, 0u, 0u};
      uint4* l4 = (uint4*)hls;
      for (int c = threadIdx.x; c < 1024; c += 256) l4[c] = z;
    } else {
#if HAVE_ASYNC_LDS
      for (int c = threadIdx.x; c < 1024; c += 256) {
        __builtin_amdgcn_global_load_async_to_lds_b128(
            (v4i_*)(h16c + c * 8), (v4i_*)(&hls[c * 8]), 0, 0);
      }
#if __has_builtin(__builtin_amdgcn_s_wait_asynccnt)
      __builtin_amdgcn_s_wait_asynccnt(0);
#else
      asm volatile("s_wait_asynccnt 0x0" ::: "memory");
#endif
#else
      uint4* l4 = (uint4*)hls;
      for (int c = threadIdx.x; c < 1024; c += 256) l4[c] = ((const uint4*)h16c)[c];
#endif
    }
    __syncthreads();

    v8f accR = {}, accZ = {}, accN = {};
    const unsigned int* hrow = (const unsigned int*)&hls[ln15 * H];
    for (int kt = 0; kt < Kt; ++kt) {
      FragU fa;
      int   k0 = kt * 32;
#pragma unroll
      for (int v = 0; v < 8; ++v) {
        int klo = (v < 4) ? (8 * half + 2 * v) : (16 + 8 * half + 2 * (v - 4));
        fa.d[v] = hrow[(k0 + klo) >> 1];     // two consecutive bf16 K-values
      }
      const uint4* b0 = (const uint4*)(whh + ((size_t)(jt)*Kt + kt) * 512 + lane * 16);
      const uint4* b1 = (const uint4*)(whh + ((size_t)(64 + jt) * Kt + kt) * 512 + lane * 16);
      const uint4* b2 = (const uint4*)(whh + ((size_t)(128 + jt) * Kt + kt) * 512 + lane * 16);
      FragU fr, fz, fn;
      fr.q[0] = b0[0]; fr.q[1] = b0[1];
      fz.q[0] = b1[0]; fz.q[1] = b1[1];
      fn.q[0] = b2[0]; fn.q[1] = b2[1];
      accR = __builtin_amdgcn_wmma_f32_16x16x32_bf16(false, fa.v, false, fr.v, (short)0, accR, false, false);
      accZ = __builtin_amdgcn_wmma_f32_16x16x32_bf16(false, fa.v, false, fz.v, (short)0, accZ, false, false);
      accN = __builtin_amdgcn_wmma_f32_16x16x32_bf16(false, fa.v, false, fn.v, (short)0, accN, false, false);
    }

    // epilogue: lanes 0-15 hold batch rows 0..7 (acc element i == batch i)
    if (half == 0) {
      int   j  = j0 + ln15;
      float br = bhh[j], bz = bhh[H + j], bn = bhh[2 * H + j];
#pragma unroll
      for (int b = 0; b < B_; ++b) {
        size_t xrow = ((size_t)b * S_ + te) * (size_t)G3_;
        float  xr   = xg[xrow + j];
        float  xz   = xg[xrow + H + j];
        float  xn   = xg[xrow + 2 * H + j];
        float  r    = sigmoidf_(xr + accR[b] + br);
        float  z    = sigmoidf_(xz + accZ[b] + bz);
        float  n    = tanhf(xn + r * (accN[b] + bn));
        float  hp   = (t == 0) ? 0.0f : hcur[b * H + j];
        float  hv   = (1.0f - z) * n + z * hp;
        hnxt[b * H + j] = hv;
        h16n[b * H + j] = f2bf(hv);
        out[((size_t)b * S_ + te) * (size_t)(2 * H) + col_off + j] = hv;
      }
    }
    __syncthreads();   // all waves done with LDS before next stage

    // grid-wide barrier between timesteps
    if (threadIdx.x == 0) {
      __threadfence();
      atomicAdd(counter, 1u);
      unsigned int target = (unsigned int)nblk * (unsigned int)(t + 1);
      while (__hip_atomic_load(counter, __ATOMIC_RELAXED, __HIP_MEMORY_SCOPE_AGENT) < target) {
        __builtin_amdgcn_s_sleep(2);
      }
      __threadfence();
    }
    __syncthreads();
  }
}

__global__ void init_counters(unsigned int* c) {
  if (threadIdx.x < 16) c[threadIdx.x] = 0u;
}

// ------------------------------------------------------------- host -------
static inline void launch_pack_a(const float* A, unsigned short* dst, int M, int K,
                                 hipStream_t s) {
  int Mt = (M + 15) / 16, Kt = K / 32;
  int total = Mt * Kt * 32;
  pack_a_bf16<<<(total + 255) / 256, 256, 0, s>>>(A, dst, M, K, Kt, total);
}
static inline void launch_pack_b(const float* W, unsigned short* dst, int N, int K,
                                 hipStream_t s) {
  int Nt = N / 16, Kt = K / 32;
  int total = Nt * Kt * 32;
  pack_b_bf16<<<(total + 255) / 256, 256, 0, s>>>(W, dst, N, K, Kt, total);
}
static inline void launch_gemm(const unsigned short* Ap, const unsigned short* Bp,
                               const float* bias, const float* resid, float* C, int M,
                               int N, int K, hipStream_t s) {
  int Mt2 = (M + 31) / 32, Nt4 = N / 64, Kt = K / 32;
  int tiles = Mt2 * Nt4;
  wmma_gemm_kernel<<<(tiles + 7) / 8, 256, 0, s>>>(Ap, Bp, bias, resid, C, Mt2, Nt4, Kt, M, N);
}

extern "C" void kernel_launch(void* const* d_in, const int* in_sizes, int n_in,
                              void* d_out, int out_size, void* d_ws, size_t ws_size,
                              hipStream_t stream) {
  (void)in_sizes; (void)n_in; (void)out_size; (void)ws_size;
  const float* x          = (const float*)d_in[0];
  const float* gru_norm_w = (const float*)d_in[1];
  const float* w_ih_l0    = (const float*)d_in[2];
  const float* w_hh_l0    = (const float*)d_in[3];
  const float* b_ih_l0    = (const float*)d_in[4];
  const float* b_hh_l0    = (const float*)d_in[5];
  const float* w_ih_l1    = (const float*)d_in[6];
  const float* w_hh_l1    = (const float*)d_in[7];
  const float* b_ih_l1    = (const float*)d_in[8];
  const float* b_hh_l1    = (const float*)d_in[9];
  const float* gru_out_w  = (const float*)d_in[10];
  const float* ffn_norm_w = (const float*)d_in[11];
  const float* w1         = (const float*)d_in[12];
  const float* w2         = (const float*)d_in[13];
  const float* w3         = (const float*)d_in[14];
  float*       outp       = (float*)d_out;

  char*  base = (char*)d_ws;
  size_t off  = 0;
  auto   alloc = [&](size_t bytes) -> char* {
    char* p = base + off;
    off += (bytes + 255) & ~(size_t)255;
    return p;
  };

  // packed bf16 weight fragments
  unsigned short* pk_ih0[2] = {(unsigned short*)alloc((size_t)G3_ * D_ * 2),
                               (unsigned short*)alloc((size_t)G3_ * D_ * 2)};
  unsigned short* pk_hh0[2] = {(unsigned short*)alloc((size_t)G3_ * H_ * 2),
                               (unsigned short*)alloc((size_t)G3_ * H_ * 2)};
  unsigned short* pk_ih1[2] = {(unsigned short*)alloc((size_t)G3_ * 2 * H_ * 2),
                               (unsigned short*)alloc((size_t)G3_ * 2 * H_ * 2)};
  unsigned short* pk_hh1[2] = {(unsigned short*)alloc((size_t)G3_ * H_ * 2),
                               (unsigned short*)alloc((size_t)G3_ * H_ * 2)};
  unsigned short* pk_out    = (unsigned short*)alloc((size_t)D_ * 2 * H_ * 2);
  unsigned short* pk_w1     = (unsigned short*)alloc((size_t)FFN_ * D_ * 2);
  unsigned short* pk_w3     = (unsigned short*)alloc((size_t)FFN_ * D_ * 2);
  unsigned short* pk_w2     = (unsigned short*)alloc((size_t)D_ * FFN_ * 2);

  float*          h0    = (float*)alloc((size_t)M_ * D_ * 4);        // rmsnorm out / x1
  unsigned short* apack = (unsigned short*)alloc((size_t)M_ * FFN_ * 2);
  float*          xg_f  = (float*)alloc((size_t)M_ * G3_ * 4);
  float*          xg_b  = (float*)alloc((size_t)M_ * G3_ * 4);
  float*          hcat  = (float*)alloc((size_t)M_ * 2 * H_ * 4);
  float*          hn    = (float*)alloc((size_t)M_ * D_ * 4);
  float*          f1    = (float*)alloc((size_t)M_ * FFN_ * 4);
  float*          f3    = (float*)alloc((size_t)M_ * FFN_ * 4);
  float*          hbuf  = (float*)alloc((size_t)2 * B_ * H_ * 4);
  unsigned short* hb16  = (unsigned short*)alloc((size_t)2 * B_ * H_ * 2);
  unsigned int*   cnt   = (unsigned int*)alloc(64);

  init_counters<<<1, 32, 0, stream>>>(cnt);

  // pack all weights once per launch
  launch_pack_b(w_ih_l0,                         pk_ih0[0], G3_, D_,     stream);
  launch_pack_b(w_ih_l0 + (size_t)G3_ * D_,      pk_ih0[1], G3_, D_,     stream);
  launch_pack_b(w_hh_l0,                         pk_hh0[0], G3_, H_,     stream);
  launch_pack_b(w_hh_l0 + (size_t)G3_ * H_,      pk_hh0[1], G3_, H_,     stream);
  launch_pack_b(w_ih_l1,                         pk_ih1[0], G3_, 2 * H_, stream);
  launch_pack_b(w_ih_l1 + (size_t)G3_ * 2 * H_,  pk_ih1[1], G3_, 2 * H_, stream);
  launch_pack_b(w_hh_l1,                         pk_hh1[0], G3_, H_,     stream);
  launch_pack_b(w_hh_l1 + (size_t)G3_ * H_,      pk_hh1[1], G3_, H_,     stream);
  launch_pack_b(gru_out_w,                       pk_out,    D_,  2 * H_, stream);
  launch_pack_b(w1,                              pk_w1,     FFN_, D_,    stream);
  launch_pack_b(w3,                              pk_w3,     FFN_, D_,    stream);
  launch_pack_b(w2,                              pk_w2,     D_,  FFN_,   stream);

  // ---- h = rmsnorm(x) ----
  rmsnorm_kernel<<<M_, 256, 0, stream>>>(x, gru_norm_w, h0, D_);

  // ---- layer 0: gate precompute + scans ----
  launch_pack_a(h0, apack, M_, D_, stream);
  launch_gemm(apack, pk_ih0[0], b_ih_l0,       nullptr, xg_f, M_, G3_, D_, stream);
  launch_gemm(apack, pk_ih0[1], b_ih_l0 + G3_, nullptr, xg_b, M_, G3_, D_, stream);
  gru_scan_kernel<<<8, 256, 0, stream>>>(xg_f, pk_hh0[0], b_hh_l0,       hbuf, hb16, hcat, cnt + 0, 0, 0);
  gru_scan_kernel<<<8, 256, 0, stream>>>(xg_b, pk_hh0[1], b_hh_l0 + G3_, hbuf, hb16, hcat, cnt + 1, 1, H_);

  // ---- layer 1 ----
  launch_pack_a(hcat, apack, M_, 2 * H_, stream);
  launch_gemm(apack, pk_ih1[0], b_ih_l1,       nullptr, xg_f, M_, G3_, 2 * H_, stream);
  launch_gemm(apack, pk_ih1[1], b_ih_l1 + G3_, nullptr, xg_b, M_, G3_, 2 * H_, stream);
  gru_scan_kernel<<<8, 256, 0, stream>>>(xg_f, pk_hh1[0], b_hh_l1,       hbuf, hb16, hcat, cnt + 2, 0, 0);
  gru_scan_kernel<<<8, 256, 0, stream>>>(xg_b, pk_hh1[1], b_hh_l1 + G3_, hbuf, hb16, hcat, cnt + 3, 1, H_);

  // ---- residual 1: x1 = x + hcat @ gru_out_w^T ----
  launch_pack_a(hcat, apack, M_, 2 * H_, stream);
  launch_gemm(apack, pk_out, nullptr, x, h0, M_, D_, 2 * H_, stream);   // h0 := x1

  // ---- FFN: SwiGLU ----
  rmsnorm_kernel<<<M_, 256, 0, stream>>>(h0, ffn_norm_w, hn, D_);
  launch_pack_a(hn, apack, M_, D_, stream);
  launch_gemm(apack, pk_w1, nullptr, nullptr, f1, M_, FFN_, D_, stream);
  launch_gemm(apack, pk_w3, nullptr, nullptr, f3, M_, FFN_, D_, stream);
  {
    // fused SwiGLU + pack for the down-projection GEMM
    int Mt = (M_ + 15) / 16, Kt = FFN_ / 32;
    int total = Mt * Kt * 32;
    pack_a_swiglu<<<(total + 255) / 256, 256, 0, stream>>>(f1, f3, apack, M_, FFN_, Kt, total);
  }
  launch_gemm(apack, pk_w2, nullptr, h0, outp, M_, D_, FFN_, stream);   // out = x1 + ffn
}
